// KNeighborSelect_29171417875197
// MI455X (gfx1250) — matrix operand order repaired and checked by reading, not compiled, not run on hardware
//
#include <hip/hip_runtime.h>

typedef __bf16 v16bf __attribute__((ext_vector_type(16)));
typedef float  v8f   __attribute__((ext_vector_type(8)));

#define B_      16
#define N_      4096
#define DP_     32
#define C_      64
#define K_      16
#define KP1     17
#define WPB     4
#define IDXMASK 0xFFFu
#define KEYMASK 0xFFFFF000u

#define NORMS_BYTES 0x40000u            // 65536 floats
#define SPLIT_BYTES 0x400000u           // B*N*DP bf16 = 4 MB each

// ---------------------------------------------------------------------------
// Pass 1a: squared norms of every point -> ws.norms
// ---------------------------------------------------------------------------
__global__ __launch_bounds__(256) void knn_norms_kernel(
    const float* __restrict__ pts, float* __restrict__ norms) {
  int tid = blockIdx.x * blockDim.x + threadIdx.x;
  const float4* p = (const float4*)(pts + (size_t)tid * DP_);
  float acc = 0.f;
#pragma unroll
  for (int i = 0; i < DP_ / 4; ++i) {
    float4 v = p[i];
    acc += v.x * v.x + v.y * v.y + v.z * v.z + v.w * v.w;
  }
  norms[tid] = acc;
}

// ---------------------------------------------------------------------------
// Pass 1b: one-time f32 -> (bf16 hi, bf16 lo) split of all points, packed
// pairs, same row-major [N][32] element order as pts. Amortizes the split
// out of every wave's 256-iteration hot loop.
// ---------------------------------------------------------------------------
__device__ __forceinline__ unsigned pack2(__bf16 a, __bf16 b) {
  return (unsigned)__builtin_bit_cast(unsigned short, a) |
         ((unsigned)__builtin_bit_cast(unsigned short, b) << 16);
}

__global__ __launch_bounds__(256) void knn_split_kernel(
    const float* __restrict__ pts, uint2* __restrict__ hi, uint2* __restrict__ lo) {
  int tid = blockIdx.x * blockDim.x + threadIdx.x;   // over B*N*DP/4
  float4 v = ((const float4*)pts)[tid];
  __bf16 h0 = (__bf16)v.x, h1 = (__bf16)v.y, h2 = (__bf16)v.z, h3 = (__bf16)v.w;
  __bf16 l0 = (__bf16)(v.x - (float)h0), l1 = (__bf16)(v.y - (float)h1);
  __bf16 l2 = (__bf16)(v.z - (float)h2), l3 = (__bf16)(v.w - (float)h3);
  hi[tid] = make_uint2(pack2(h0, h1), pack2(h2, h3));
  lo[tid] = make_uint2(pack2(l0, l1), pack2(l2, l3));
}

// Packed-key sorted insert: key = (f32 distance bits & ~0xFFF) | index.
// Ascending list; each step is one v_min_u32 + v_max_u32 compare-exchange.
__device__ __forceinline__ void key_insert(unsigned key, unsigned (&hk)[KP1]) {
  if (key < hk[KP1 - 1]) {
    unsigned ck = key;
#pragma unroll
    for (int j = 0; j < KP1; ++j) {
      unsigned lo = min(ck, hk[j]);
      unsigned hi = max(ck, hk[j]);
      hk[j] = lo;
      ck = hi;
    }
  }
}

__device__ __forceinline__ unsigned mkkey(float d, unsigned idx) {
  return (__float_as_uint(fmaxf(d, 0.0f)) & KEYMASK) | idx;
}

struct U32x8 { uint4 a, b; };
__device__ __forceinline__ v16bf cat16(uint4 a, uint4 b) {
  U32x8 t{a, b};
  return __builtin_bit_cast(v16bf, t);
}

// ---------------------------------------------------------------------------
// Pass 2: fused distances (bf16-split WMMA, A = candidates so C lands
// lane-per-query-row, no LDS transpose) -> packed-key top-17 -> gather.
// One wave owns 16 query rows and scans all 256 candidate tiles.
// ---------------------------------------------------------------------------
__global__ __launch_bounds__(32 * WPB) void knn_edge_kernel(
    const char* __restrict__ phi, const char* __restrict__ plo,
    const float* __restrict__ fts, const float* __restrict__ norms,
    float* __restrict__ out) {
  const int lane = threadIdx.x & 31;
  const int wv   = threadIdx.x >> 5;
  const int tile = blockIdx.x * WPB + wv;           // 0 .. B*(N/16)-1
  const int b    = tile >> 8;
  const int n0   = (tile & 255) << 4;
  const int hw   = lane >> 4;
  const int ln   = lane & 15;

  const char*  hiB    = phi + (size_t)b * N_ * DP_ * 2;
  const char*  loB    = plo + (size_t)b * N_ * DP_ * 2;
  const float* normsB = norms + b * N_;

  // ---- B operand: THE QUERIES (K x 16, N = query) -- loop-invariant -----
  // lane%16 = N (query row); per-lane K = 16*hw + 0..15 (contiguous, 32 B)
  v16bf b_hi, b_lo;
  {
    size_t qoff = ((size_t)(n0 + ln) * DP_ + 16 * hw) * 2;
    b_hi = cat16(*(const uint4*)(hiB + qoff), *(const uint4*)(hiB + qoff + 16));
    b_lo = cat16(*(const uint4*)(loB + qoff), *(const uint4*)(loB + qoff + 16));
  }
  const float rn = normsB[n0 + ln];                 // this lane's query norm

  unsigned hk[KP1];
#pragma unroll
  for (int j = 0; j < KP1; ++j) hk[j] = 0xFFFFFFFFu;

  for (int mt = 0; mt < N_ / 16; ++mt) {
    const int m0 = mt << 4;

    // ---- A operand: THE CANDIDATES (16 x K, M = candidate) --------------
    // lane%16 = M; per-lane K = 16*hw + 0..15 (contiguous, matches B)
    size_t aoff = ((size_t)(m0 + ln) * DP_ + 16 * hw) * 2;
    v16bf a_hi = cat16(*(const uint4*)(hiB + aoff), *(const uint4*)(hiB + aoff + 16));
    v16bf a_lo = cat16(*(const uint4*)(loB + aoff), *(const uint4*)(loB + aoff + 16));
    if (mt + 1 < N_ / 16)
      __builtin_prefetch(hiB + aoff + (size_t)16 * DP_ * 2, 0, 1);

    // dot(P_m, P_n) ~= hi*hi + lo*hi + hi*lo  (f32 accumulate)
    // C layout: lane ln = query, VGPR v + 8*hw = candidate (m0 + v + 8*hw)
    v8f c = {};
    c = __builtin_amdgcn_wmma_f32_16x16x32_bf16(false, a_hi, false, b_hi, (short)0, c, false, false);
    c = __builtin_amdgcn_wmma_f32_16x16x32_bf16(false, a_lo, false, b_hi, (short)0, c, false, false);
    c = __builtin_amdgcn_wmma_f32_16x16x32_bf16(false, a_hi, false, b_lo, (short)0, c, false, false);

    // candidate norms for m0+8*hw .. +7 (broadcast within half-wave)
    const float4 cn0 = *(const float4*)(normsB + m0 + 8 * hw);
    const float4 cn1 = *(const float4*)(normsB + m0 + 8 * hw + 4);
    const unsigned mb = (unsigned)(m0 + 8 * hw);

    key_insert(mkkey(fmaf(-2.0f, c[0], rn + cn0.x), mb + 0), hk);
    key_insert(mkkey(fmaf(-2.0f, c[1], rn + cn0.y), mb + 1), hk);
    key_insert(mkkey(fmaf(-2.0f, c[2], rn + cn0.z), mb + 2), hk);
    key_insert(mkkey(fmaf(-2.0f, c[3], rn + cn0.w), mb + 3), hk);
    key_insert(mkkey(fmaf(-2.0f, c[4], rn + cn1.x), mb + 4), hk);
    key_insert(mkkey(fmaf(-2.0f, c[5], rn + cn1.y), mb + 5), hk);
    key_insert(mkkey(fmaf(-2.0f, c[6], rn + cn1.z), mb + 6), hk);
    key_insert(mkkey(fmaf(-2.0f, c[7], rn + cn1.w), mb + 7), hk);
  }

  // ---- merge half-wave partner lists (snapshot first, then insert) ------
  unsigned pk[KP1];
#pragma unroll
  for (int j = 0; j < KP1; ++j)
    pk[j] = (unsigned)__shfl_xor((int)hk[j], 16, 32);
  if (lane < 16) {
#pragma unroll
    for (int j = 0; j < KP1; ++j) key_insert(pk[j], hk);
  }
  // lanes 0..15: sorted top-17 keys for row (n0+lane); hk[0] = self.

  // ---- gather + write: one coalesced 512B b128 store per (row, nbr) -----
  const float* fbase = fts + (size_t)b * N_ * C_;
  const int   ch  = ln * 4;                  // feature channel chunk
  const float sel = (float)hw;               // 0 -> copy, 1 -> knn - center

  for (int row = 0; row < 16; ++row) {
    float4 c4 = *(const float4*)(fbase + (size_t)(n0 + row) * C_ + ch);
    size_t obase = (size_t)(b * N_ + n0 + row) * K_ * (2 * C_);
#pragma unroll
    for (int j = 0; j < K_; ++j) {
      unsigned m = ((unsigned)__shfl((int)hk[j + 1], row, 32)) & IDXMASK;
      float4 f4 = *(const float4*)(fbase + (size_t)m * C_ + ch);
      float4 o4;
      o4.x = fmaf(-sel, c4.x, f4.x);
      o4.y = fmaf(-sel, c4.y, f4.y);
      o4.z = fmaf(-sel, c4.z, f4.z);
      o4.w = fmaf(-sel, c4.w, f4.w);
      *(float4*)(out + obase + (size_t)j * (2 * C_) + lane * 4) = o4;
    }
  }
}

// ---------------------------------------------------------------------------
extern "C" void kernel_launch(void* const* d_in, const int* in_sizes, int n_in,
                              void* d_out, int out_size, void* d_ws, size_t ws_size,
                              hipStream_t stream) {
  (void)in_sizes; (void)n_in; (void)out_size; (void)ws_size;
  const float* pts = (const float*)d_in[0];
  const float* fts = (const float*)d_in[1];
  // d_in[2] is k (==16), compile-time specialized.
  float* norms = (float*)d_ws;                              // 256 KB
  char*  phi   = (char*)d_ws + NORMS_BYTES;                 // 4 MB bf16 hi
  char*  plo   = (char*)d_ws + NORMS_BYTES + SPLIT_BYTES;   // 4 MB bf16 lo
  float* out   = (float*)d_out;

  knn_norms_kernel<<<(B_ * N_) / 256, 256, 0, stream>>>(pts, norms);
  knn_split_kernel<<<(B_ * N_ * DP_ / 4) / 256, 256, 0, stream>>>(
      pts, (uint2*)phi, (uint2*)plo);
  knn_edge_kernel<<<(B_ * (N_ / 16)) / WPB, 32 * WPB, 0, stream>>>(
      phi, plo, fts, norms, out);
}
